// DualMemoryAttention_33483565040104
// MI455X (gfx1250) — matrix operand order repaired and checked by reading, not compiled, not run on hardware
//
#include <hip/hip_runtime.h>

typedef __bf16  v16bf __attribute__((ext_vector_type(16)));
typedef float   v8f   __attribute__((ext_vector_type(8)));
typedef float   v2f   __attribute__((ext_vector_type(2)));

#define BT       2048      // B*T
#define D_MODEL  256
#define DVAL     128
#define DEMB     4
#define NLTM     32768
#define NSTM     2048
#define DKLTM    64
#define DKSTM    16
#define INF_F    3.402823466e38f

__device__ __forceinline__ unsigned short f2bf(float f) {
    unsigned u = __float_as_uint(f);
    u += 0x7FFFu + ((u >> 16) & 1u);          // round-to-nearest-even
    return (unsigned short)(u >> 16);
}
__device__ __forceinline__ float sigmoidf_(float x) { return 1.f / (1.f + __expf(-x)); }

union BFrag {
    unsigned short s[16];
    uint4          u[2];
    v16bf          v;
};

// ---------------------------------------------------------------------------
// Kernel 0: pack center keys to bf16 and precompute ||k||^2
// ---------------------------------------------------------------------------
template<int N, int DK>
__global__ __launch_bounds__(64)
void pack_norm_kernel(const float* __restrict__ K,
                      unsigned int* __restrict__ kb,   // 2 bf16 per uint
                      float* __restrict__ norms)
{
    int row = blockIdx.x * 64 + threadIdx.x;
    if (row >= N) return;
    float nrm = 0.f;
#pragma unroll 4
    for (int k = 0; k < DK; k += 2) {
        float a = K[(size_t)row * DK + k];
        float b = K[(size_t)row * DK + k + 1];
        nrm += a * a + b * b;
        kb[((size_t)row * DK + k) >> 1] = (unsigned)f2bf(a) | ((unsigned)f2bf(b) << 16);
    }
    norms[row] = nrm;
}

// ---------------------------------------------------------------------------
// Kernel 1: memory read. One block = 16 queries, 8 waves scan all centers.
// Scores via v_wmma_f32_16x16x32_bf16; per-lane register top-k; LDS merge;
// RBF weights (literal reference semantics incl. +1e-8); gather V/E.
// ---------------------------------------------------------------------------
template<int DK, int TOPK, int NC>
__global__ __launch_bounds__(256)
void read_mem_kernel(const float* __restrict__ qv,
                     const uint4* __restrict__ kb,
                     const float* __restrict__ norms,
                     const float* __restrict__ Vm,
                     const float* __restrict__ Em,
                     float* __restrict__ rV,
                     float* __restrict__ rE)
{
    constexpr int NFRAG = (DK + 31) / 32;
    constexpr int CAND  = 16 * TOPK;

    __shared__ float          sS[8 * 256];       // per-wave 16x16 score tiles
    __shared__ float          mS[16 * CAND];     // merge candidates: scores
    __shared__ unsigned short mI[16 * CAND];     // merge candidates: center ids

    const int tid  = threadIdx.x;
    const int wave = tid >> 5;
    const int lane = tid & 31;
    const int hl   = lane >> 4;                  // lane half
    const int ln   = lane & 15;
    const int qbase = blockIdx.x * 16;

    // --- A fragments (queries, bf16, 16 contiguous K per lane half) ---
    BFrag af[NFRAG];
#pragma unroll
    for (int j = 0; j < NFRAG; j++) {
#pragma unroll
        for (int e = 0; e < 16; e++) {
            int k = j * 32 + hl * 16 + e;
            float fv = (k < DK) ? qv[(size_t)(qbase + ln) * DK + k] : 0.f;
            af[j].s[e] = f2bf(fv);
        }
    }

    // --- per-lane sorted top-k (lane covers query ln, center half hl) ---
    float lst[TOPK]; int li[TOPK];
#pragma unroll
    for (int t = 0; t < TOPK; t++) { lst[t] = INF_F; li[t] = 0; }

    for (int tile = wave; tile < NC / 16; tile += 8) {
        if (tile + 8 < NC / 16)
            __builtin_prefetch((const void*)(kb + (((size_t)((tile + 8) * 16 + ln) * DK) >> 3)), 0, 1);

        v8f acc = {0.f, 0.f, 0.f, 0.f, 0.f, 0.f, 0.f, 0.f};
#pragma unroll
        for (int j = 0; j < NFRAG; j++) {
            BFrag bf;
            int kbase = j * 32 + hl * 16;        // B: 16 contiguous K per lane half
            if (kbase + 16 <= DK) {
                const uint4* p = kb + (((size_t)(tile * 16 + ln) * DK + kbase) >> 3);
                bf.u[0] = p[0]; bf.u[1] = p[1];
            } else {
                bf.u[0] = make_uint4(0u, 0u, 0u, 0u);
                bf.u[1] = make_uint4(0u, 0u, 0u, 0u);
            }
            acc = __builtin_amdgcn_wmma_f32_16x16x32_bf16(false, af[j].v, false, bf.v,
                                                          (short)0, acc, false, false);
        }
        float kk = norms[tile * 16 + ln];
        // D layout: element v -> (M = v + 8*hl, N = ln). Transpose via LDS.
#pragma unroll
        for (int v = 0; v < 8; v++)
            sS[wave * 256 + ln * 16 + (v + 8 * hl)] = kk - 2.f * acc[v];

        // each lane consumes 8 centers of its query's row (same-wave LDS: in order)
#pragma unroll
        for (int i = 0; i < 8; i++) {
            int c = hl * 8 + i;
            float s = sS[wave * 256 + c * 16 + ln];
            int  ci = tile * 16 + c;
            if (s < lst[TOPK - 1]) {
                float cv = s; int cvi = ci;
#pragma unroll
                for (int t = 0; t < TOPK; t++) {
                    bool sw = cv < lst[t];
                    float a = lst[t]; int b = li[t];
                    lst[t] = sw ? cv : a;  li[t] = sw ? cvi : b;
                    cv = sw ? a : cv;      cvi = sw ? b : cvi;
                }
            }
        }
    }

    // --- dump partial lists to LDS merge buffer ---
    {
        int slot = wave * 2 + hl;                // 16 lists per query
#pragma unroll
        for (int t = 0; t < TOPK; t++) {
            mS[ln * CAND + slot * TOPK + t] = lst[t];
            mI[ln * CAND + slot * TOPK + t] = (unsigned short)li[t];
        }
    }
    __syncthreads();

    // --- phase B: wave w finalizes queries 2w, 2w+1 ---
    for (int sub = 0; sub < 2; ++sub) {
        int qi = wave * 2 + sub;
        int qg = qbase + qi;

        float qn = 0.f;                          // ||q||^2 (needed: eps breaks cancellation)
        for (int k = lane; k < DK; k += 32) { float v = qv[(size_t)qg * DK + k]; qn += v * v; }
#pragma unroll
        for (int off = 16; off >= 1; off >>= 1) qn += __shfl_xor(qn, off, 32);

        const int base = qi * CAND;
        float selW[TOPK]; int selI[TOPK]; float ssum = 0.f;
#pragma unroll
        for (int r = 0; r < TOPK; r++) {
            float mv = INF_F; int mp = 0;
            for (int i = lane; i < CAND; i += 32) {
                float s = mS[base + i];
                if (s < mv) { mv = s; mp = i; }
            }
#pragma unroll
            for (int off = 16; off >= 1; off >>= 1) {
                float ov = __shfl_xor(mv, off, 32);
                int   op = __shfl_xor(mp, off, 32);
                if (ov < mv || (ov == mv && op < mp)) { mv = ov; mp = op; }
            }
            selI[r] = mI[base + mp];
            float w = __expf(-0.5f * (qn + mv)); // exp(-d2 / (2*sigma^2)), sigma=1
            selW[r] = w; ssum += w;
            if (lane == 0) mS[base + mp] = INF_F;
        }
        float inv = 1.f / (ssum + 1e-8f);

        float a0 = 0.f, a1 = 0.f, a2 = 0.f, a3 = 0.f;
#pragma unroll
        for (int r = 0; r < TOPK; r++) {
            const float4* vp = (const float4*)(Vm + (size_t)selI[r] * DVAL);
            float4 vv = vp[lane];
            float w = selW[r];
            a0 += w * vv.x; a1 += w * vv.y; a2 += w * vv.z; a3 += w * vv.w;
        }
        float4 o; o.x = a0 * inv; o.y = a1 * inv; o.z = a2 * inv; o.w = a3 * inv;
        ((float4*)(rV + (size_t)qg * DVAL))[lane] = o;

        if (lane < DEMB) {
            float e = 0.f;
#pragma unroll
            for (int r = 0; r < TOPK; r++) e += selW[r] * Em[(size_t)selI[r] * DEMB + lane];
            rE[(size_t)qg * DEMB + lane] = e * inv;
        }
    }
}

// ---------------------------------------------------------------------------
// Kernel 2a: gates + blend (wave per query)
// ---------------------------------------------------------------------------
__global__ __launch_bounds__(256)
void gate_blend_kernel(const float* __restrict__ x,
                       const float* __restrict__ g_prior,
                       const float* __restrict__ rVl, const float* __restrict__ rEl,
                       const float* __restrict__ rVs, const float* __restrict__ rEs,
                       const float* __restrict__ Wgl, const float* __restrict__ bgl,
                       const float* __restrict__ gpwl,
                       const float* __restrict__ Wgs, const float* __restrict__ bgs,
                       const float* __restrict__ gpws,
                       const float* __restrict__ lwp, const float* __restrict__ swp,
                       float* __restrict__ rVb, float* __restrict__ outE,
                       float* __restrict__ outG)
{
    int wave = threadIdx.x >> 5, lane = threadIdx.x & 31;
    int qg = blockIdx.x * 8 + wave;

    float dl = 0.f, ds = 0.f;
    for (int k = lane; k < D_MODEL; k += 32) {
        float xv = x[(size_t)qg * D_MODEL + k];
        dl += xv * Wgl[k]; ds += xv * Wgs[k];
    }
    for (int k = lane; k < DVAL; k += 32) {
        dl += rVl[(size_t)qg * DVAL + k] * Wgl[D_MODEL + k];
        ds += rVs[(size_t)qg * DVAL + k] * Wgs[D_MODEL + k];
    }
#pragma unroll
    for (int off = 16; off >= 1; off >>= 1) {
        dl += __shfl_xor(dl, off, 32);
        ds += __shfl_xor(ds, off, 32);
    }
    float lw = sigmoidf_(lwp[0]), sw = sigmoidf_(swp[0]);
    float itw = 1.f / (lw + sw + 1e-8f);
    float gp = g_prior[qg];
    float gl = sigmoidf_(dl + bgl[0] + gpwl[0] * gp);
    float gs = sigmoidf_(ds + bgs[0] + gpws[0] * gp);
    if (lane == 0) outG[qg] = (lw * gl + sw * gs) * itw;
    for (int k = lane; k < DVAL; k += 32)
        rVb[(size_t)qg * DVAL + k] =
            (lw * rVl[(size_t)qg * DVAL + k] + sw * rVs[(size_t)qg * DVAL + k]) * itw;
    if (lane < DEMB)
        outE[(size_t)qg * DEMB + lane] =
            (lw * rEl[(size_t)qg * DEMB + lane] + sw * rEs[(size_t)qg * DEMB + lane]) * itw;
}

// ---------------------------------------------------------------------------
// Kernel 2b: M = r_V @ Wout + bout via v_wmma_f32_16x16x4_f32 (fp32 semantics)
// ---------------------------------------------------------------------------
__global__ __launch_bounds__(256)
void out_gemm_kernel(const float* __restrict__ rVb,
                     const float* __restrict__ Wout,
                     const float* __restrict__ bout,
                     float* __restrict__ M)
{
    int wave = threadIdx.x >> 5, lane = threadIdx.x & 31;
    int hl = lane >> 4, ln = lane & 15;
    int tile = blockIdx.x * 8 + wave;            // 128 M-tiles x 16 N-tiles
    int mt = tile >> 4, nt = tile & 15;

    v8f acc = {0.f, 0.f, 0.f, 0.f, 0.f, 0.f, 0.f, 0.f};
    for (int k0 = 0; k0 < DVAL; k0 += 4) {
        int ka = k0 + 2 * hl;                    // A/B f32: element v -> K = 2*hl + v
        v2f a = {0.f, 0.f}, b = {0.f, 0.f};
        a[0] = rVb[(size_t)(mt * 16 + ln) * DVAL + ka];
        a[1] = rVb[(size_t)(mt * 16 + ln) * DVAL + ka + 1];
        b[0] = Wout[(size_t)(ka)     * D_MODEL + nt * 16 + ln];
        b[1] = Wout[(size_t)(ka + 1) * D_MODEL + nt * 16 + ln];
        acc = __builtin_amdgcn_wmma_f32_16x16x4_f32(false, a, false, b,
                                                    (short)0, acc, false, false);
    }
#pragma unroll
    for (int v = 0; v < 8; v++) {
        int row = mt * 16 + v + 8 * hl;
        int col = nt * 16 + ln;
        M[(size_t)row * D_MODEL + col] = acc[v] + bout[col];
    }
}

// ---------------------------------------------------------------------------
extern "C" void kernel_launch(void* const* d_in, const int* in_sizes, int n_in,
                              void* d_out, int out_size, void* d_ws, size_t ws_size,
                              hipStream_t stream)
{
    (void)in_sizes; (void)n_in; (void)out_size; (void)ws_size;

    const float* x       = (const float*)d_in[0];
    const float* q_ltm   = (const float*)d_in[1];
    const float* q_stm   = (const float*)d_in[2];
    const float* g_prior = (const float*)d_in[3];
    const float* ltm_K   = (const float*)d_in[4];
    const float* ltm_V   = (const float*)d_in[5];
    const float* ltm_E   = (const float*)d_in[6];
    const float* stm_K   = (const float*)d_in[7];
    const float* stm_V   = (const float*)d_in[8];
    const float* stm_E   = (const float*)d_in[9];
    const float* Wg_l    = (const float*)d_in[10];
    const float* bg_l    = (const float*)d_in[11];
    const float* gpw_l   = (const float*)d_in[12];
    const float* Wg_s    = (const float*)d_in[13];
    const float* bg_s    = (const float*)d_in[14];
    const float* gpw_s   = (const float*)d_in[15];
    const float* lww     = (const float*)d_in[16];
    const float* sww     = (const float*)d_in[17];
    const float* Wout    = (const float*)d_in[18];
    const float* bout    = (const float*)d_in[19];

    char* ws = (char*)d_ws;
    size_t off = 0;
    auto alloc = [&](size_t bytes) -> void* {
        void* p = ws + off;
        off = (off + bytes + 255) & ~(size_t)255;
        return p;
    };
    unsigned* kb_l = (unsigned*)alloc((size_t)NLTM * DKLTM * 2);
    unsigned* kb_s = (unsigned*)alloc((size_t)NSTM * DKSTM * 2);
    float* nrm_l = (float*)alloc((size_t)NLTM * 4);
    float* nrm_s = (float*)alloc((size_t)NSTM * 4);
    float* rV_l  = (float*)alloc((size_t)BT * DVAL * 4);
    float* rE_l  = (float*)alloc((size_t)BT * DEMB * 4);
    float* rV_s  = (float*)alloc((size_t)BT * DVAL * 4);
    float* rE_s  = (float*)alloc((size_t)BT * DEMB * 4);
    float* rV_b  = (float*)alloc((size_t)BT * DVAL * 4);

    float* outM = (float*)d_out;
    float* outE = outM + (size_t)BT * D_MODEL;
    float* outG = outE + (size_t)BT * DEMB;

    pack_norm_kernel<NLTM, DKLTM><<<NLTM / 64, 64, 0, stream>>>(ltm_K, kb_l, nrm_l);
    pack_norm_kernel<NSTM, DKSTM><<<NSTM / 64, 64, 0, stream>>>(stm_K, kb_s, nrm_s);

    read_mem_kernel<DKLTM, 32, NLTM><<<BT / 16, 256, 0, stream>>>(
        q_ltm, (const uint4*)kb_l, nrm_l, ltm_V, ltm_E, rV_l, rE_l);
    read_mem_kernel<DKSTM, 16, NSTM><<<BT / 16, 256, 0, stream>>>(
        q_stm, (const uint4*)kb_s, nrm_s, stm_V, stm_E, rV_s, rE_s);

    gate_blend_kernel<<<BT / 8, 256, 0, stream>>>(
        x, g_prior, rV_l, rE_l, rV_s, rE_s,
        Wg_l, bg_l, gpw_l, Wg_s, bg_s, gpw_s, lww, sww,
        rV_b, outE, outG);

    out_gemm_kernel<<<(BT / 16) * 16 / 8, 256, 0, stream>>>(rV_b, Wout, bout, outM);
}
